// MCANet_91250875171209
// MI455X (gfx1250) — matrix operand order repaired and checked by reading, not compiled, not run on hardware
//
#include <hip/hip_runtime.h>

// ---------- types for WMMA (gfx1250, wave32) ----------
typedef __attribute__((ext_vector_type(16))) __bf16   v16bf;
typedef __attribute__((ext_vector_type(8)))  float    v8f;
typedef __attribute__((ext_vector_type(4)))  unsigned u32x4;

#define Bsz 32
#define LD  512
#define LP  4096
#define Hd  128

// encode float -> unsigned preserving order (for atomicMax-based float max)
__device__ __forceinline__ unsigned fenc(float f) {
    unsigned u = __float_as_uint(f);
    return (u & 0x80000000u) ? ~u : (u | 0x80000000u);
}
__device__ __forceinline__ float fdec(unsigned u) {
    return __uint_as_float((u & 0x80000000u) ? (u ^ 0x80000000u) : ~u);
}
#define ENC_NEG_MAX 0x00800000u   // fenc(-FLT_MAX)

// fp32 -> bf16 round-to-nearest-even
__device__ __forceinline__ unsigned short f2bf(float v) {
    unsigned u = __float_as_uint(v);
    u += 0x7FFFu + ((u >> 16) & 1u);
    return (unsigned short)(u >> 16);
}
__device__ __forceinline__ unsigned pack2(float a, float b) {
    return (unsigned)f2bf(a) | ((unsigned)f2bf(b) << 16);
}

// ---------------- init row-max accumulators ----------------
__global__ __launch_bounds__(256) void k_init_rmax(unsigned* __restrict__ rmax, int n) {
    int i = blockIdx.x * 256 + threadIdx.x;
    if (i < n) rmax[i] = ENC_NEG_MAX;
}

// ---------------- fused gather + bf16 WMMA GEMM + row/col max ----------------
// grid: (8 m-chunks, 32 batches), block: 256 threads = 8 waves (wave32)
__global__ __launch_bounds__(256) void k_aff_max(
    const int*   __restrict__ drug_ids,   // [B,LD]
    const int*   __restrict__ prot_ids,   // [B,LP]
    const float* __restrict__ drug_emb,   // [600,H]
    const float* __restrict__ prot_emb,   // [26,H]
    unsigned*    __restrict__ rmax,       // [B*LD] encoded
    float*       __restrict__ colmax)     // [B*LP]
{
    __shared__ unsigned short sA[128 * 128];   // 32 KB: l-block  (bf16)
    __shared__ unsigned short sB[64 * 128];    // 16 KB: m-block  (bf16)
    __shared__ unsigned scolmax[512];          // 2 KB : per-chunk col max (encoded)

    const int tid  = threadIdx.x;
    const int b    = blockIdx.y;
    const int mc   = blockIdx.x;          // m-chunk: 512 columns
    const int lane = tid & 31;
    const int wv   = tid >> 5;            // wave 0..7
    const int hf   = lane >> 4;           // lane half (0/1)
    const int ln   = lane & 15;

    for (int i = tid; i < 512; i += 256) scolmax[i] = ENC_NEG_MAX;

    for (int lb = 0; lb < 4; ++lb) {      // 4 l-blocks of 128 rows
        const int l0 = lb * 128;
        __syncthreads();
        // stage sA: gather d_feat rows [l0..l0+127] and convert to bf16
        for (int k = 0; k < 8; ++k) {
            int idx = tid + k * 256;          // 0..2047 u32x4 slots
            int row = idx >> 4;               // 16 slots per 128-col row
            int seg = idx & 15;               // 8 bf16 per slot
            int id  = drug_ids[b * LD + l0 + row];
            const float* src = drug_emb + (size_t)id * Hd + seg * 8;
            u32x4 o;
            o.x = pack2(src[0], src[1]);
            o.y = pack2(src[2], src[3]);
            o.z = pack2(src[4], src[5]);
            o.w = pack2(src[6], src[7]);
            ((u32x4*)sA)[idx] = o;
        }
        __syncthreads();

        // A fragments for this wave's 16-row l-tile (ISA 16-bit A 16x32 layout)
        const int lr = wv * 16;
        union { u32x4 q[2]; v16bf v; } afr[4];
        {
            const unsigned short* rp = &sA[(lr + ln) * 128];
#pragma unroll
            for (int kc = 0; kc < 4; ++kc) {
                int k0 = kc * 32;
                afr[kc].q[0] = *((const u32x4*)(rp + k0 + hf * 8));        // i=0..7
                afr[kc].q[1] = *((const u32x4*)(rp + k0 + 16 + hf * 8));   // i=8..15
            }
        }
        // deferred per-lane row-max partials: rm[r] covers columns == ln (mod 16)
        float rm[8];
#pragma unroll
        for (int r = 0; r < 8; ++r) rm[r] = -3.4e38f;

        for (int mb = 0; mb < 8; ++mb) {   // 8 m-blocks of 64 rows
            __syncthreads();
            // stage sB: gather p_feat rows for this m-block, convert to bf16
            for (int k = 0; k < 4; ++k) {
                int idx = tid + k * 256;       // 0..1023
                int row = idx >> 4;            // 0..63
                int seg = idx & 15;
                int id  = prot_ids[b * LP + mc * 512 + mb * 64 + row];
                const float* src = prot_emb + (size_t)id * Hd + seg * 8;
                u32x4 o;
                o.x = pack2(src[0], src[1]);
                o.y = pack2(src[2], src[3]);
                o.z = pack2(src[4], src[5]);
                o.w = pack2(src[6], src[7]);
                ((u32x4*)sB)[idx] = o;
            }
            __syncthreads();

            // two m-tiles per iteration: two independent WMMA accumulator chains
#pragma unroll
            for (int mp = 0; mp < 2; ++mp) {
                const int mt0 = mp * 2, mt1 = mp * 2 + 1;
                const unsigned short* rp0 = &sB[(mt0 * 16 + ln) * 128];
                const unsigned short* rp1 = &sB[(mt1 * 16 + ln) * 128];
                union { u32x4 q[2]; v16bf v; } b0[4], b1[4];
#pragma unroll
                for (int kc = 0; kc < 4; ++kc) {
                    int k0 = kc * 32 + hf * 8;     // B layout: N=ln, K=k0..k0+7 and k0+16..k0+23
                    b0[kc].q[0] = *((const u32x4*)(rp0 + k0));
                    b0[kc].q[1] = *((const u32x4*)(rp0 + k0 + 16));
                    b1[kc].q[0] = *((const u32x4*)(rp1 + k0));
                    b1[kc].q[1] = *((const u32x4*)(rp1 + k0 + 16));
                }
                v8f c0, c1;
#pragma unroll
                for (int i = 0; i < 8; ++i) { c0[i] = 0.0f; c1[i] = 0.0f; }
#pragma unroll
                for (int kc = 0; kc < 4; ++kc) {
                    c0 = __builtin_amdgcn_wmma_f32_16x16x32_bf16(
                            false, afr[kc].v, false, b0[kc].v, (short)0, c0, false, false);
                    c1 = __builtin_amdgcn_wmma_f32_16x16x32_bf16(
                            false, afr[kc].v, false, b1[kc].v, (short)0, c1, false, false);
                }
                // C layout: VGPR r -> row M=hf*8+r, col N=ln
                float cv0 = -3.4e38f, cv1 = -3.4e38f;
#pragma unroll
                for (int r = 0; r < 8; ++r) {
                    rm[r] = fmaxf(rm[r], fmaxf(c0[r], c1[r]));  // deferred row-max
                    cv0   = fmaxf(cv0, c0[r]);
                    cv1   = fmaxf(cv1, c1[r]);
                }
                cv0 = fmaxf(cv0, __shfl_xor(cv0, 16, 32));
                cv1 = fmaxf(cv1, __shfl_xor(cv1, 16, 32));
                if (hf == 0) {
                    atomicMax(&scolmax[mb * 64 + mt0 * 16 + ln], fenc(cv0));
                    atomicMax(&scolmax[mb * 64 + mt1 * 16 + ln], fenc(cv1));
                }
            }
        }
        // cross-lane row-max reduce once per l-block, then global atomic flush
#pragma unroll
        for (int r = 0; r < 8; ++r) {
            float rv = rm[r];
            rv = fmaxf(rv, __shfl_xor(rv, 1, 32));
            rv = fmaxf(rv, __shfl_xor(rv, 2, 32));
            rv = fmaxf(rv, __shfl_xor(rv, 4, 32));
            rv = fmaxf(rv, __shfl_xor(rv, 8, 32));
            if (ln == 0)
                atomicMax(&rmax[b * LD + l0 + lr + hf * 8 + r], fenc(rv));
        }
    }
    __syncthreads();
    for (int i = tid; i < 512; i += 256)
        colmax[(size_t)b * LP + mc * 512 + i] = fdec(scolmax[i]);
}

// ---------------- softmaxes, attention pooling, MLP head ----------------
__global__ __launch_bounds__(256) void k_finalize(
    const int*   __restrict__ drug_ids, const int* __restrict__ prot_ids,
    const float* __restrict__ drug_emb, const float* __restrict__ prot_emb,
    const unsigned* __restrict__ rmax,  const float* __restrict__ colmax,
    const float* __restrict__ W1, const float* __restrict__ b1,
    const float* __restrict__ W2, const float* __restrict__ b2,
    float* __restrict__ out)
{
    __shared__ float swd[LD];
    __shared__ float swp[LP];
    __shared__ float sred[256];
    __shared__ float scomb[256];
    __shared__ float sh1[64];
    const int tid = threadIdx.x;
    const int b   = blockIdx.x;

    for (int i = tid; i < LD; i += 256) swd[i] = fdec(rmax[b * LD + i]);
    for (int i = tid; i < LP; i += 256) swp[i] = colmax[(size_t)b * LP + i];
    __syncthreads();

    // softmax over swd[512]
    float lm = -3.4e38f;
    for (int i = tid; i < LD; i += 256) lm = fmaxf(lm, swd[i]);
    sred[tid] = lm; __syncthreads();
    for (int s = 128; s > 0; s >>= 1) { if (tid < s) sred[tid] = fmaxf(sred[tid], sred[tid + s]); __syncthreads(); }
    float mx = sred[0]; __syncthreads();
    float ls = 0.f;
    for (int i = tid; i < LD; i += 256) { float e = expf(swd[i] - mx); swd[i] = e; ls += e; }
    sred[tid] = ls; __syncthreads();
    for (int s = 128; s > 0; s >>= 1) { if (tid < s) sred[tid] += sred[tid + s]; __syncthreads(); }
    float inv = 1.0f / sred[0]; __syncthreads();
    for (int i = tid; i < LD; i += 256) swd[i] *= inv;

    // softmax over swp[4096]
    __syncthreads();
    lm = -3.4e38f;
    for (int i = tid; i < LP; i += 256) lm = fmaxf(lm, swp[i]);
    sred[tid] = lm; __syncthreads();
    for (int s = 128; s > 0; s >>= 1) { if (tid < s) sred[tid] = fmaxf(sred[tid], sred[tid + s]); __syncthreads(); }
    mx = sred[0]; __syncthreads();
    ls = 0.f;
    for (int i = tid; i < LP; i += 256) { float e = expf(swp[i] - mx); swp[i] = e; ls += e; }
    sred[tid] = ls; __syncthreads();
    for (int s = 128; s > 0; s >>= 1) { if (tid < s) sred[tid] += sred[tid + s]; __syncthreads(); }
    inv = 1.0f / sred[0]; __syncthreads();
    for (int i = tid; i < LP; i += 256) swp[i] *= inv;
    __syncthreads();

    // attention-weighted pooled features (exact fp32 gather from tables)
    if (tid < 128) {
        float acc = 0.f;
        for (int l = 0; l < LD; ++l)
            acc += swd[l] * drug_emb[(size_t)drug_ids[b * LD + l] * Hd + tid];
        scomb[tid] = acc;
    } else {
        int h = tid - 128;
        float acc = 0.f;
        for (int m = 0; m < LP; ++m)
            acc += swp[m] * prot_emb[(size_t)prot_ids[b * LP + m] * Hd + h];
        scomb[128 + h] = acc;
    }
    __syncthreads();

    // MLP: 256 -> 64 (ReLU) -> 1
    if (tid < 64) {
        float a = b1[tid];
        for (int i = 0; i < 256; ++i) a += scomb[i] * W1[i * 64 + tid];
        a = fmaxf(a, 0.f);
        sh1[tid] = a * W2[tid];
    }
    __syncthreads();
    if (tid == 0) {
        float s = b2[0];
        for (int j = 0; j < 64; ++j) s += sh1[j];
        out[b] = s;
    }
}

extern "C" void kernel_launch(void* const* d_in, const int* in_sizes, int n_in,
                              void* d_out, int out_size, void* d_ws, size_t ws_size,
                              hipStream_t stream) {
    (void)in_sizes; (void)n_in; (void)out_size; (void)ws_size;
    const int*   drug_ids = (const int*)d_in[0];
    const int*   prot_ids = (const int*)d_in[1];
    const float* drug_emb = (const float*)d_in[2];
    const float* prot_emb = (const float*)d_in[3];
    const float* W1 = (const float*)d_in[4];
    const float* b1 = (const float*)d_in[5];
    const float* W2 = (const float*)d_in[6];
    const float* b2 = (const float*)d_in[7];
    float* out = (float*)d_out;

    char* ws = (char*)d_ws;
    unsigned* rmax   = (unsigned*)ws;                 // 64 KB
    float*    colmax = (float*)(ws + 65536);          // 512 KB

    k_init_rmax<<<64, 256, 0, stream>>>(rmax, Bsz * LD);
    k_aff_max<<<dim3(8, Bsz), 256, 0, stream>>>(drug_ids, prot_ids, drug_emb, prot_emb,
                                                rmax, colmax);
    k_finalize<<<Bsz, 256, 0, stream>>>(drug_ids, prot_ids, drug_emb, prot_emb,
                                        rmax, colmax, W1, b1, W2, b2, out);
}
